// Transformer_59184649339092
// MI455X (gfx1250) — matrix operand order repaired and checked by reading, not compiled, not run on hardware
//
#include <hip/hip_runtime.h>
#include <hip/hip_bf16.h>
#include <cmath>

// ---------------------------------------------------------------------------
// Model constants (match reference)
// ---------------------------------------------------------------------------
static const int H_   = 1024;
static const int NH_  = 8;
static const int L_   = 6;
static const int B_   = 4;
static const int SX_  = 512;
static const int SY_  = 512;
static const int VOUT_= 32000;
static const int DH_  = 128;     // H / NH
static const int FF_  = 4096;    // 4*H

typedef __attribute__((ext_vector_type(16))) _Float16 v16h;
typedef __attribute__((ext_vector_type(8)))  _Float16 v8h;
typedef __attribute__((ext_vector_type(4)))  _Float16 v4h;
typedef __attribute__((ext_vector_type(8)))  float    v8f;

// ---------------------------------------------------------------------------
// WMMA GEMM:  C[M,N] = act( A[M,K](fp32) @ Bh[N][K](f16, pre-transposed)
//                           + bias )
// Block tile 64(M) x 128(N), K step 64 (two WMMA sub-steps), double-buffered
// LDS -> 8 v_wmma per wave per barrier. 256 threads = 8 waves (2x4 grid),
// each wave owns a 2x2 grid of 16x16 C tiles with fragment reuse.
//
// B staging uses CDNA5 async global->LDS copies (GLOBAL_LOAD_ASYNC_TO_LDS_
// B128, ASYNCcnt): both sides are contiguous 64B chunks, INST_OFFSET applies
// to both addresses, so one LDS-offset VGPR + one 64-bit global address pair
// covers 4 chunks. s_wait_asynccnt 0 before the publishing barrier.
//
// Batched addressing: z=blockIdx.z; zo=z/zdiv; zi=z%zdiv;
//   ptr += zo*outer + zi*inner (per A,B,C).
// Requirements (satisfied here): M%64==0, N%128==0, K%64==0.
// ---------------------------------------------------------------------------
#define BM 64
#define BN 128
#define BK 64

union FragAB { v16h v; v8h h[2]; };

__device__ inline void asyncCopy64B(const _Float16* gsrc, _Float16* ldst)
{
    const unsigned lds_off = (unsigned)(uintptr_t)ldst;   // addr[31:0] = LDS offset
    asm volatile(
        "global_load_async_to_lds_b128 %0, %1, off\n\t"
        "global_load_async_to_lds_b128 %0, %1, off offset:16\n\t"
        "global_load_async_to_lds_b128 %0, %1, off offset:32\n\t"
        "global_load_async_to_lds_b128 %0, %1, off offset:48"
        :: "v"(lds_off), "v"(gsrc) : "memory");
}

__device__ inline void asyncWait()
{
    asm volatile("s_wait_asynccnt 0x0" ::: "memory");
}

__global__ __launch_bounds__(256) void gemm_wmma_kernel(
    const float* __restrict__ A, const _Float16* __restrict__ Bh,
    float* __restrict__ C, const float* __restrict__ bias,
    int M, int N, int K, int lda, int ldbh, int ldc, int act,
    long long zAo, long long zAi, long long zBo, long long zBi,
    long long zCo, long long zCi, int zdiv)
{
    __shared__ _Float16 lsA[2][BM * BK];   // [m][k]  8KB per buffer
    __shared__ _Float16 lsB[2][BN * BK];   // [n][k] 16KB per buffer

    const int z  = blockIdx.z;
    const int zo = z / zdiv;
    const int zi = z - zo * zdiv;
    A  += zo * zAo + zi * zAi;
    Bh += zo * zBo + zi * zBi;
    C  += zo * zCo + zi * zCi;

    const int tid  = threadIdx.x;
    const int wave = tid >> 5;
    const int lane = tid & 31;
    const int half = lane >> 4;     // 0 or 1
    const int l16  = lane & 15;
    const int wm   = wave >> 2;     // 0..1
    const int wn   = wave & 3;      // 0..3
    const int rowBase = blockIdx.y * BM;
    const int colBase = blockIdx.x * BN;

    // staging thread mapping
    const int am  = tid >> 2;            // 0..63
    const int ak  = (tid & 3) << 4;      // 0,16,32,48
    const int bn  = tid >> 1;            // 0..127
    const int bkk = (tid & 1) << 5;      // 0,32

    v8f acc00 = {}, acc01 = {}, acc10 = {}, acc11 = {};

    auto stage = [&](int kt, int buf) {
        const int kBase = kt * BK;
        // ---- B tile 128x64 f16: async global->LDS, no VGPR round-trip ----
        asyncCopy64B(Bh + (size_t)(colBase + bn) * ldbh + kBase + bkk,
                     &lsB[buf][bn * BK + bkk]);
        // ---- A tile 64x64 fp32 -> f16 (VALU convert overlaps async B) ----
        {
            const float* ap = A + (size_t)(rowBase + am) * lda + kBase + ak;
            const float4 f0 = *(const float4*)(ap);
            const float4 f1 = *(const float4*)(ap + 4);
            const float4 f2 = *(const float4*)(ap + 8);
            const float4 f3 = *(const float4*)(ap + 12);
            v8h h0, h1;
            h0[0] = (_Float16)f0.x; h0[1] = (_Float16)f0.y;
            h0[2] = (_Float16)f0.z; h0[3] = (_Float16)f0.w;
            h0[4] = (_Float16)f1.x; h0[5] = (_Float16)f1.y;
            h0[6] = (_Float16)f1.z; h0[7] = (_Float16)f1.w;
            h1[0] = (_Float16)f2.x; h1[1] = (_Float16)f2.y;
            h1[2] = (_Float16)f2.z; h1[3] = (_Float16)f2.w;
            h1[4] = (_Float16)f3.x; h1[5] = (_Float16)f3.y;
            h1[6] = (_Float16)f3.z; h1[7] = (_Float16)f3.w;
            *(v8h*)&lsA[buf][am * BK + ak]     = h0;
            *(v8h*)&lsA[buf][am * BK + ak + 8] = h1;
            __builtin_prefetch(ap + BK, 0, 1);    // next A tile
        }
    };

    auto compute = [&](int buf, int kofs) {
        FragAB a0, a1, b0, b1;
        const v8h* pa0 = (const v8h*)&lsA[buf][(size_t)(wm * 32 + l16) * BK + kofs];
        const v8h* pa1 = (const v8h*)&lsA[buf][(size_t)(wm * 32 + 16 + l16) * BK + kofs];
        a0.h[0] = pa0[half];  a0.h[1] = pa0[2 + half];
        a1.h[0] = pa1[half];  a1.h[1] = pa1[2 + half];
        const v8h* pb0 = (const v8h*)&lsB[buf][(size_t)(wn * 32 + l16) * BK + kofs];
        const v8h* pb1 = (const v8h*)&lsB[buf][(size_t)(wn * 32 + 16 + l16) * BK + kofs];
        b0.h[0] = pb0[half]; b0.h[1] = pb0[2 + half];
        b1.h[0] = pb1[half]; b1.h[1] = pb1[2 + half];

        acc00 = __builtin_amdgcn_wmma_f32_16x16x32_f16(
            false, a0.v, false, b0.v, (short)0, acc00, false, false);
        acc01 = __builtin_amdgcn_wmma_f32_16x16x32_f16(
            false, a0.v, false, b1.v, (short)0, acc01, false, false);
        acc10 = __builtin_amdgcn_wmma_f32_16x16x32_f16(
            false, a1.v, false, b0.v, (short)0, acc10, false, false);
        acc11 = __builtin_amdgcn_wmma_f32_16x16x32_f16(
            false, a1.v, false, b1.v, (short)0, acc11, false, false);
    };

    const int nk = K / BK;
    stage(0, 0);
    asyncWait();
    __syncthreads();
    for (int kt = 0; kt < nk; ++kt) {
        const int cur = kt & 1;
        if (kt + 1 < nk) stage(kt + 1, cur ^ 1);
        compute(cur, 0);
        compute(cur, 32);
        asyncWait();          // own async copies into other buffer done
        __syncthreads();      // publish to all waves
    }

    // ---- epilogue: lane l16 = N col; VGPR r -> M = half*8 + r ----
    const int gcol0 = colBase + wn * 32 + l16;
    const int gcol1 = gcol0 + 16;
    const float bv0 = bias ? bias[gcol0] : 0.0f;
    const float bv1 = bias ? bias[gcol1] : 0.0f;
#pragma unroll
    for (int r = 0; r < 8; ++r) {
        const int grow0 = rowBase + wm * 32 + half * 8 + r;
        const int grow1 = grow0 + 16;
        float v00 = acc00[r] + bv0, v01 = acc01[r] + bv1;
        float v10 = acc10[r] + bv0, v11 = acc11[r] + bv1;
        if (act) {
            v00 = v00 > 0.0f ? v00 : 0.0f; v01 = v01 > 0.0f ? v01 : 0.0f;
            v10 = v10 > 0.0f ? v10 : 0.0f; v11 = v11 > 0.0f ? v11 : 0.0f;
        }
        __builtin_nontemporal_store(v00, &C[(size_t)grow0 * ldc + gcol0]);
        __builtin_nontemporal_store(v01, &C[(size_t)grow0 * ldc + gcol1]);
        __builtin_nontemporal_store(v10, &C[(size_t)grow1 * ldc + gcol0]);
        __builtin_nontemporal_store(v11, &C[(size_t)grow1 * ldc + gcol1]);
    }
}

// ---------------------------------------------------------------------------
// fp32 [R][C] -> f16 [C][R] convert + transpose (32x32 LDS tile, batched)
// ---------------------------------------------------------------------------
__global__ __launch_bounds__(256) void cvt_transpose_kernel(
    const float* __restrict__ src, _Float16* __restrict__ dst,
    int ldsrc, int lddst,
    long long zSo, long long zSi, long long zDo, long long zDi, int zdiv)
{
    __shared__ float tile[32][33];
    const int z  = blockIdx.z;
    const int zo = z / zdiv;
    const int zi = z - zo * zdiv;
    src += zo * zSo + zi * zSi;
    dst += zo * zDo + zi * zDi;

    const int tx = threadIdx.x & 31;
    const int ty = threadIdx.x >> 5;      // 0..7
    const int r0 = blockIdx.y * 32;
    const int c0 = blockIdx.x * 32;
#pragma unroll
    for (int j = 0; j < 4; ++j)
        tile[ty + 8 * j][tx] = src[(size_t)(r0 + ty + 8 * j) * ldsrc + c0 + tx];
    __syncthreads();
#pragma unroll
    for (int j = 0; j < 4; ++j)
        dst[(size_t)(c0 + ty + 8 * j) * lddst + r0 + tx] =
            (_Float16)tile[tx][ty + 8 * j];
}

// fp32 -> f16 flat convert (n % 1024 == 0)
__global__ __launch_bounds__(256) void cvt_f16_kernel(
    const float* __restrict__ src, _Float16* __restrict__ dst)
{
    const size_t i = ((size_t)blockIdx.x * 256 + threadIdx.x) * 4;
    const float4 f = *(const float4*)(src + i);
    v4h h;
    h[0] = (_Float16)f.x; h[1] = (_Float16)f.y;
    h[2] = (_Float16)f.z; h[3] = (_Float16)f.w;
    *(v4h*)(dst + i) = h;
}

// ---------------------------------------------------------------------------
// Elementwise helpers
// ---------------------------------------------------------------------------
__device__ inline float waveSum(float v) {
    for (int o = 16; o > 0; o >>= 1) v += __shfl_down(v, o, 32);
    return v;
}
__device__ inline float waveMax(float v) {
    for (int o = 16; o > 0; o >>= 1) v = fmaxf(v, __shfl_down(v, o, 32));
    return v;
}

// out[b,s,:] = emb[ids[b,s],:] + posenc(s,:)
__global__ __launch_bounds__(256) void embed_pe_kernel(
    const float* __restrict__ emb, const int* __restrict__ ids,
    float* __restrict__ out, int S, int Hd)
{
    const int s = blockIdx.x;
    const int b = blockIdx.y;
    const int id = ids[b * S + s];
    const float* e = emb + (size_t)id * Hd;
    float* o = out + ((size_t)b * S + s) * Hd;
    const float ln1e4 = 9.210340371976184f;   // log(10000)
    for (int j = threadIdx.x; j < Hd; j += 256) {
        const int i = j >> 1;
        const float ang = (float)s * __expf(-((float)i / (float)Hd) * ln1e4);
        const float pe = (j & 1) ? __cosf(ang) : __sinf(ang);
        o[j] = e[j] + pe;
    }
}

// masked scaled softmax over last dim of S[B,NH,Sq,Sk] (in place)
// causal==0: valid keys k < xlen[b];  causal==1: valid keys k <= q
__global__ __launch_bounds__(128) void softmax_kernel(
    float* __restrict__ S, const int* __restrict__ xlen,
    int causal, int Sq, int Sk, float scale)
{
    const int q = blockIdx.x;
    const int h = blockIdx.y;
    const int b = blockIdx.z;
    float* row = S + (((size_t)(b * gridDim.y + h) * Sq) + q) * (size_t)Sk;
    const int lim = causal ? (q + 1) : xlen[b];
    const int tid = threadIdx.x;
    __shared__ float red[4];

    float mx = -INFINITY;
    for (int k = tid; k < Sk; k += 128) {
        const float v = (k < lim) ? row[k] * scale : -INFINITY;
        mx = fmaxf(mx, v);
    }
    mx = waveMax(mx);
    if ((tid & 31) == 0) red[tid >> 5] = mx;
    __syncthreads();
    mx = fmaxf(fmaxf(red[0], red[1]), fmaxf(red[2], red[3]));
    __syncthreads();

    float sum = 0.0f;
    for (int k = tid; k < Sk; k += 128)
        if (k < lim) sum += __expf(row[k] * scale - mx);
    sum = waveSum(sum);
    if ((tid & 31) == 0) red[tid >> 5] = sum;
    __syncthreads();
    sum = red[0] + red[1] + red[2] + red[3];
    const float inv = 1.0f / sum;

    for (int k = tid; k < Sk; k += 128)
        row[k] = (k < lim) ? __expf(row[k] * scale - mx) * inv : 0.0f;
}

// out = LayerNorm(x + r) * g + b   (one block per row; out may alias x)
__global__ __launch_bounds__(256) void add_ln_kernel(
    const float* x, const float* r,
    const float* __restrict__ g, const float* __restrict__ bta,
    float* out, int Hd)
{
    const size_t row = blockIdx.x;
    const float* xr = x + row * Hd;
    const float* rr = r + row * Hd;
    float* o = out + row * Hd;
    const int tid = threadIdx.x;
    __shared__ float rs[8], rs2[8];

    float s = 0.0f, s2 = 0.0f;
    for (int j = tid; j < Hd; j += 256) {
        const float v = xr[j] + rr[j];
        s += v; s2 += v * v;
    }
    s = waveSum(s); s2 = waveSum(s2);
    if ((tid & 31) == 0) { rs[tid >> 5] = s; rs2[tid >> 5] = s2; }
    __syncthreads();
    s = 0.0f; s2 = 0.0f;
#pragma unroll
    for (int w = 0; w < 8; ++w) { s += rs[w]; s2 += rs2[w]; }
    const float mean = s / (float)Hd;
    const float var  = s2 / (float)Hd - mean * mean;
    const float inv  = rsqrtf(var + 1e-5f);

    for (int j = tid; j < Hd; j += 256) {
        const float v = xr[j] + rr[j];
        o[j] = (v - mean) * inv * g[j] + bta[j];
    }
}

// in-place log-softmax over V per row
__global__ __launch_bounds__(256) void logsoftmax_kernel(
    float* __restrict__ logits, int V)
{
    const size_t row = blockIdx.x;
    float* p = logits + row * (size_t)V;
    const int tid = threadIdx.x;
    __shared__ float red[8];

    float mx = -INFINITY;
    for (int j = tid; j < V; j += 256) mx = fmaxf(mx, p[j]);
    mx = waveMax(mx);
    if ((tid & 31) == 0) red[tid >> 5] = mx;
    __syncthreads();
#pragma unroll
    for (int w = 0; w < 8; ++w) mx = fmaxf(mx, red[w]);
    __syncthreads();

    float sum = 0.0f;
    for (int j = tid; j < V; j += 256) sum += __expf(p[j] - mx);
    sum = waveSum(sum);
    if ((tid & 31) == 0) red[tid >> 5] = sum;
    __syncthreads();
    sum = 0.0f;
#pragma unroll
    for (int w = 0; w < 8; ++w) sum += red[w];
    const float lse = mx + __logf(sum);

    for (int j = tid; j < V; j += 256) p[j] = p[j] - lse;
}

// ---------------------------------------------------------------------------
// Host-side launch helpers
// ---------------------------------------------------------------------------
static void launch_gemm(hipStream_t st, const float* A, const _Float16* Bh,
                        float* C, const float* bias, int M, int N, int K,
                        int lda, int ldbh, int ldc, int act,
                        int zcount = 1, int zdiv = 1,
                        long long zAo = 0, long long zAi = 0,
                        long long zBo = 0, long long zBi = 0,
                        long long zCo = 0, long long zCi = 0)
{
    dim3 grid(N / BN, M / BM, zcount), block(256);
    gemm_wmma_kernel<<<grid, block, 0, st>>>(A, Bh, C, bias, M, N, K,
                                             lda, ldbh, ldc, act,
                                             zAo, zAi, zBo, zBi, zCo, zCi, zdiv);
}

// fp32 [R][C] (ldsrc) -> f16 [C][R] (lddst)
static void launch_cvtT(hipStream_t st, const float* src, _Float16* dst,
                        int R, int C, int ldsrc, int lddst,
                        int zcount = 1, int zdiv = 1,
                        long long zSo = 0, long long zSi = 0,
                        long long zDo = 0, long long zDi = 0)
{
    dim3 grid(C / 32, R / 32, zcount), block(256);
    cvt_transpose_kernel<<<grid, block, 0, st>>>(src, dst, ldsrc, lddst,
                                                 zSo, zSi, zDo, zDi, zdiv);
}

// Multi-head attention: q_in [B,Sq,H], kv_in [B,Skv,H] -> OUT [B,Sq,H]
static void run_mha(hipStream_t st,
                    const float* q_in, const float* kv_in, int Sq, int Skv,
                    const float* wq, const float* wk, const float* wv,
                    const float* wo, int causal, const int* xlen,
                    float* QB, float* KB, float* VB, float* SC, float* CTX,
                    float* OUT, _Float16* KB16, _Float16* V16T, _Float16* WB16)
{
    // projections (weights converted+transposed to f16 [N][K] first)
    launch_cvtT(st, wq, WB16, H_, H_, H_, H_);
    launch_gemm(st, q_in,  WB16, QB, nullptr, B_ * Sq,  H_, H_, H_, H_, H_, 0);
    launch_cvtT(st, wk, WB16, H_, H_, H_, H_);
    launch_gemm(st, kv_in, WB16, KB, nullptr, B_ * Skv, H_, H_, H_, H_, H_, 0);
    launch_cvtT(st, wv, WB16, H_, H_, H_, H_);
    launch_gemm(st, kv_in, WB16, VB, nullptr, B_ * Skv, H_, H_, H_, H_, H_, 0);

    // K-matrix to f16 (already [key][DH] per head = [N][K] slices)
    cvt_f16_kernel<<<dim3((B_ * Skv * H_) / 1024), dim3(256), 0, st>>>(KB, KB16);
    // scores: per (b,h)  S = Q @ K^T   [Sq,Skv]
    launch_gemm(st, QB, KB16, SC, nullptr, Sq, Skv, DH_, H_, H_, Skv, 0,
                B_ * NH_, NH_,
                (long long)Sq * H_,  DH_,
                (long long)Skv * H_, DH_,
                (long long)NH_ * Sq * Skv, (long long)Sq * Skv);
    // softmax
    const float scale = 1.0f / sqrtf((float)DH_);
    softmax_kernel<<<dim3(Sq, NH_, B_), dim3(128), 0, st>>>(
        SC, xlen, causal, Sq, Skv, scale);

    // V^T per (b,h): fp32 [Skv][DH] (ld H) -> f16 [DH][Skv]
    launch_cvtT(st, VB, V16T, Skv, DH_, H_, Skv,
                B_ * NH_, NH_,
                (long long)Skv * H_, DH_,
                (long long)NH_ * DH_ * Skv, (long long)DH_ * Skv);
    // context: per (b,h)  C = S @ V   [Sq,DH]
    launch_gemm(st, SC, V16T, CTX, nullptr, Sq, DH_, Skv, Skv, Skv, H_, 0,
                B_ * NH_, NH_,
                (long long)NH_ * Sq * Skv, (long long)Sq * Skv,
                (long long)NH_ * DH_ * Skv, (long long)DH_ * Skv,
                (long long)Sq * H_, DH_);
    // output projection
    launch_cvtT(st, wo, WB16, H_, H_, H_, H_);
    launch_gemm(st, CTX, WB16, OUT, nullptr, B_ * Sq, H_, H_, H_, H_, H_, 0);
}

// ---------------------------------------------------------------------------
// Full model
// ---------------------------------------------------------------------------
extern "C" void kernel_launch(void* const* d_in, const int* in_sizes, int n_in,
                              void* d_out, int out_size, void* d_ws, size_t ws_size,
                              hipStream_t stream)
{
    (void)in_sizes; (void)n_in; (void)out_size; (void)ws_size;

    const int*   x_ids   = (const int*)  d_in[0];
    const int*   x_len   = (const int*)  d_in[1];
    const int*   y_ids   = (const int*)  d_in[2];
    const float* emb_enc = (const float*)d_in[3];
    const float* emb_dec = (const float*)d_in[4];
    const float* enc_wq  = (const float*)d_in[5];
    const float* enc_wk  = (const float*)d_in[6];
    const float* enc_wv  = (const float*)d_in[7];
    const float* enc_wo  = (const float*)d_in[8];
    const float* enc_g1  = (const float*)d_in[9];
    const float* enc_b1  = (const float*)d_in[10];
    const float* enc_w1  = (const float*)d_in[11];
    const float* enc_bb1 = (const float*)d_in[12];
    const float* enc_w2  = (const float*)d_in[13];
    const float* enc_bb2 = (const float*)d_in[14];
    const float* enc_g2  = (const float*)d_in[15];
    const float* enc_b2  = (const float*)d_in[16];
    const float* dec_wq1 = (const float*)d_in[17];
    const float* dec_wk1 = (const float*)d_in[18];
    const float* dec_wv1 = (const float*)d_in[19];
    const float* dec_wo1 = (const float*)d_in[20];
    const float* dec_g1  = (const float*)d_in[21];
    const float* dec_b1  = (const float*)d_in[22];
    const float* dec_wq2 = (const float*)d_in[23];
    const float* dec_wk2 = (const float*)d_in[24];
    const float* dec_wv2 = (const float*)d_in[25];
    const float* dec_wo2 = (const float*)d_in[26];
    const float* dec_g2  = (const float*)d_in[27];
    const float* dec_b2  = (const float*)d_in[28];
    const float* dec_w1  = (const float*)d_in[29];
    const float* dec_bb1 = (const float*)d_in[30];
    const float* dec_w2  = (const float*)d_in[31];
    const float* dec_bb2 = (const float*)d_in[32];
    const float* dec_g3  = (const float*)d_in[33];
    const float* dec_b3  = (const float*)d_in[34];
    const float* gen_w   = (const float*)d_in[35];
    const float* gen_b   = (const float*)d_in[36];

    // workspace bump allocator (fp32 element units)
    float* base = (float*)d_ws;
    size_t off = 0;
    auto alloc = [&](size_t n) { float* p = base + off; off += n; return p; };
    const size_t tokE = (size_t)B_ * SX_ * H_;   // 2M floats
    float* ZENC = alloc(tokE);
    float* HDEC = alloc(tokE);
    float* QB   = alloc(tokE);
    float* KB   = alloc(tokE);
    float* VB   = alloc(tokE);
    float* SC   = alloc((size_t)B_ * NH_ * SX_ * SX_);  // 8M floats
    float* CTX  = alloc(tokE);
    float* T1   = alloc((size_t)B_ * SX_ * FF_);        // 8M floats
    float* T2   = alloc(tokE);
    _Float16* KB16 = (_Float16*)alloc(tokE / 2);                   // B*S*H f16
    _Float16* V16T = (_Float16*)alloc(tokE / 2);                   // B*S*H f16
    _Float16* WB16 = (_Float16*)alloc(((size_t)H_ * VOUT_) / 2);   // max weight f16

    // ---- encoder ----
    embed_pe_kernel<<<dim3(SX_, B_), dim3(256), 0, stream>>>(
        emb_enc, x_ids, ZENC, SX_, H_);

    for (int l = 0; l < L_; ++l) {
        const size_t wOff  = (size_t)l * H_ * H_;
        const size_t fOff1 = (size_t)l * H_ * FF_;
        const size_t fOff2 = (size_t)l * FF_ * H_;
        const size_t vOff  = (size_t)l * H_;
        const size_t fvOff = (size_t)l * FF_;

        run_mha(stream, ZENC, ZENC, SX_, SX_,
                enc_wq + wOff, enc_wk + wOff, enc_wv + wOff, enc_wo + wOff,
                /*causal=*/0, x_len, QB, KB, VB, SC, CTX, T2, KB16, V16T, WB16);
        add_ln_kernel<<<dim3(B_ * SX_), dim3(256), 0, stream>>>(
            ZENC, T2, enc_g1 + vOff, enc_b1 + vOff, ZENC, H_);

        launch_cvtT(stream, enc_w1 + fOff1, WB16, H_, FF_, FF_, H_);
        launch_gemm(stream, ZENC, WB16, T1, enc_bb1 + fvOff,
                    B_ * SX_, FF_, H_, H_, H_, FF_, /*relu*/1);
        launch_cvtT(stream, enc_w2 + fOff2, WB16, FF_, H_, H_, FF_);
        launch_gemm(stream, T1, WB16, T2, enc_bb2 + vOff,
                    B_ * SX_, H_, FF_, FF_, FF_, H_, 0);
        add_ln_kernel<<<dim3(B_ * SX_), dim3(256), 0, stream>>>(
            ZENC, T2, enc_g2 + vOff, enc_b2 + vOff, ZENC, H_);
    }

    // ---- decoder ----
    embed_pe_kernel<<<dim3(SY_, B_), dim3(256), 0, stream>>>(
        emb_dec, y_ids, HDEC, SY_, H_);

    for (int l = 0; l < L_; ++l) {
        const size_t wOff  = (size_t)l * H_ * H_;
        const size_t fOff1 = (size_t)l * H_ * FF_;
        const size_t fOff2 = (size_t)l * FF_ * H_;
        const size_t vOff  = (size_t)l * H_;
        const size_t fvOff = (size_t)l * FF_;

        // causal self-attention
        run_mha(stream, HDEC, HDEC, SY_, SY_,
                dec_wq1 + wOff, dec_wk1 + wOff, dec_wv1 + wOff, dec_wo1 + wOff,
                /*causal=*/1, nullptr, QB, KB, VB, SC, CTX, T2, KB16, V16T, WB16);
        add_ln_kernel<<<dim3(B_ * SY_), dim3(256), 0, stream>>>(
            HDEC, T2, dec_g1 + vOff, dec_b1 + vOff, HDEC, H_);

        // cross-attention over encoder memory
        run_mha(stream, HDEC, ZENC, SY_, SX_,
                dec_wq2 + wOff, dec_wk2 + wOff, dec_wv2 + wOff, dec_wo2 + wOff,
                /*causal=*/0, x_len, QB, KB, VB, SC, CTX, T2, KB16, V16T, WB16);
        add_ln_kernel<<<dim3(B_ * SY_), dim3(256), 0, stream>>>(
            HDEC, T2, dec_g2 + vOff, dec_b2 + vOff, HDEC, H_);

        // FFN
        launch_cvtT(stream, dec_w1 + fOff1, WB16, H_, FF_, FF_, H_);
        launch_gemm(stream, HDEC, WB16, T1, dec_bb1 + fvOff,
                    B_ * SY_, FF_, H_, H_, H_, FF_, /*relu*/1);
        launch_cvtT(stream, dec_w2 + fOff2, WB16, FF_, H_, H_, FF_);
        launch_gemm(stream, T1, WB16, T2, dec_bb2 + vOff,
                    B_ * SY_, H_, FF_, FF_, FF_, H_, 0);
        add_ln_kernel<<<dim3(B_ * SY_), dim3(256), 0, stream>>>(
            HDEC, T2, dec_g3 + vOff, dec_b3 + vOff, HDEC, H_);
    }

    // ---- generator + log-softmax (in place in d_out) ----
    float* logits = (float*)d_out;
    launch_cvtT(stream, gen_w, WB16, H_, VOUT_, VOUT_, H_);
    launch_gemm(stream, HDEC, WB16, logits, gen_b,
                B_ * SY_, VOUT_, H_, H_, H_, VOUT_, 0);
    logsoftmax_kernel<<<dim3(B_ * SY_), dim3(256), 0, stream>>>(logits, VOUT_);
}
